// HeterogeneousAttributeEncoder_46179488366834
// MI455X (gfx1250) — matrix operand order repaired and checked by reading, not compiled, not run on hardware
//
#include <hip/hip_runtime.h>
#include <hip/hip_bf16.h>

typedef __attribute__((ext_vector_type(16))) _Float16 v16h;
typedef __attribute__((ext_vector_type(8)))  float    v8f;

#define D 128
#define TOKENS_PER_BLOCK 128
#define NTOK (32 * 1024)

__device__ __forceinline__ unsigned pack_h2(float a, float b) {
    union { _Float16 h[2]; unsigned u; } p;
    p.h[0] = (_Float16)a;
    p.h[1] = (_Float16)b;
    return p.u;
}

__global__ __launch_bounds__(256)
void hae_encoder_kernel(const float* __restrict__ ais,
                        const float* __restrict__ W_lon1, const float* __restrict__ b_lon1,
                        const float* __restrict__ W_lon2, const float* __restrict__ b_lon2,
                        const float* __restrict__ W_lat1, const float* __restrict__ b_lat1,
                        const float* __restrict__ W_lat2, const float* __restrict__ b_lat2,
                        const float* __restrict__ W_cog,  const float* __restrict__ b_cog,
                        const float* __restrict__ W_head, const float* __restrict__ b_head,
                        const float* __restrict__ W_dr,   const float* __restrict__ b_dr,
                        const float* __restrict__ W_sog,  const float* __restrict__ b_sog,
                        const float* __restrict__ W_rot,  const float* __restrict__ b_rot,
                        const float* __restrict__ emb,
                        float* __restrict__ out)
{
    // Pre-swizzled B fragments for the two 128x128 GEMMs:
    // [coord][kt(4)][nt(8)][lane(32)][j(8 dwords = 16 halves)]  = 64 KB
    __shared__ unsigned w2frag[2][4][8][32][8];

    const int tid = threadIdx.x;

    // ---- cooperative fill of B fragments (W2 f32 -> f16, fragment order) ----
    // B (32x16) layout assumed: lane n = lane&15 selects column N; VGPR j holds
    // K = 16*(j/4)+2*(j%4) (+8 for lanes 16-31), matching A.
    for (int i = 0; i < 64; ++i) {
        int idx   = tid + 256 * i;          // 0 .. 16383
        int coord = idx >> 13;
        int kt    = (idx >> 11) & 3;
        int nt    = (idx >> 8) & 7;
        int lane  = (idx >> 3) & 31;
        int j     = idx & 7;
        int k0    = kt * 32 + 16 * (j >> 2) + 2 * (j & 3) + 8 * (lane >> 4);
        int n     = nt * 16 + (lane & 15);
        const float* W2 = coord ? W_lat2 : W_lon2;
        w2frag[coord][kt][nt][lane][j] = pack_h2(W2[k0 * D + n], W2[(k0 + 1) * D + n]);
    }
    __syncthreads();

    const int lane = tid & 31;
    const int wave = tid >> 5;
    const int tile = blockIdx.x * TOKENS_PER_BLOCK + wave * 16;   // 16 tokens per wave
    const float deg2rad = 0.017453292519943295f;

    // =====================  lon / lat paths (WMMA)  =====================
    for (int coord = 0; coord < 2; ++coord) {
        const float* W1 = coord ? W_lat1 : W_lon1;
        const float* b1 = coord ? b_lat1 : b_lon1;
        const float* b2 = coord ? b_lat2 : b_lon2;
        const float maskval = coord ? 91.0f : 181.0f;

        // token owned by this lane for the A fragment
        float x  = ais[(size_t)(tile + (lane & 15)) * 8 + coord];
        float xr = x * deg2rad;
        float sv = __sinf(xr);
        float cv = __cosf(xr);

        // Build 4 A fragments (16x32 f16 each) entirely in registers.
        // A layout: lanes 0-15 = M, VGPR j holds K = 16*(j/4)+2*(j%4) (+8 for lanes 16-31).
        v16h a[4];
        #pragma unroll
        for (int kt = 0; kt < 4; ++kt) {
            #pragma unroll
            for (int j = 0; j < 8; ++j) {
                int k0 = kt * 32 + 16 * (j >> 2) + 2 * (j & 3) + 8 * (lane >> 4);
                float h0 = sv * W1[k0]     + cv * W1[D + k0]     + b1[k0];
                float h1 = sv * W1[k0 + 1] + cv * W1[D + k0 + 1] + b1[k0 + 1];
                h0 = h0 > 0.0f ? h0 : 0.0f;
                h1 = h1 > 0.0f ? h1 : 0.0f;
                a[kt][2 * j]     = (_Float16)h0;
                a[kt][2 * j + 1] = (_Float16)h1;
            }
        }

        // Per-lane mask for the 8 output rows this lane's C fragment covers.
        float mask8[8];
        #pragma unroll
        for (int r = 0; r < 8; ++r) {
            int t = tile + r + 8 * (lane >> 4);
            float xv = ais[(size_t)t * 8 + coord];
            mask8[r] = (xv == maskval) ? 0.0f : 1.0f;
        }

        const int n = lane & 15;
        for (int nt = 0; nt < 8; ++nt) {
            float b2v = b2[nt * 16 + n];
            v8f c;
            #pragma unroll
            for (int r = 0; r < 8; ++r) c[r] = b2v;   // C init = bias (row-invariant)

            #pragma unroll
            for (int kt = 0; kt < 4; ++kt) {
                const v16h* bp =
                    reinterpret_cast<const v16h*>(&w2frag[coord][kt][nt][lane][0]);
                v16h bfrag = *bp;                     // 2x ds_load_b128
                c = __builtin_amdgcn_wmma_f32_16x16x32_f16(
                        /*neg_a=*/false, a[kt],
                        /*neg_b=*/false, bfrag,
                        /*c_mod=*/(short)0, c,
                        /*reuse_a=*/false, /*reuse_b=*/false);
            }

            #pragma unroll
            for (int r = 0; r < 8; ++r) {
                int t = tile + r + 8 * (lane >> 4);
                out[(size_t)t * 1024 + coord * D + nt * 16 + n] = c[r] * mask8[r];
            }
        }
    }

    // =====================  cheap encoders (VALU)  =====================
    // slots: 2=rot 3=sog 4=cog 5=head 6=vtype 7=dr
    for (int it = 0; it < 64; ++it) {
        int m = it >> 2;                    // token within tile
        int d = ((it & 3) << 5) + lane;     // dim 0..127, coalesced per wave
        int t = tile + m;
        size_t base  = (size_t)t * 8;
        size_t obase = (size_t)t * 1024;

        float rot  = ais[base + 2];
        float sog  = ais[base + 3];
        float cog  = ais[base + 4];
        float head = ais[base + 5];
        float vt   = ais[base + 6];
        float dr   = ais[base + 7];

        // rot: continuous in [-127, 127]
        {
            float xn = 2.0f * (rot + 127.0f) * (1.0f / 254.0f) - 1.0f;
            float e  = tanhf(xn * W_rot[d] + b_rot[d]);
            out[obase + 2 * D + d] = (rot == -1.0f) ? 0.0f : e;
        }
        // sog: continuous in [0, 40]
        {
            float xn = 2.0f * sog * (1.0f / 40.0f) - 1.0f;
            float e  = tanhf(xn * W_sog[d] + b_sog[d]);
            out[obase + 3 * D + d] = (sog == -1.0f) ? 0.0f : e;
        }
        // cog: cyclical
        {
            float xr = cog * deg2rad - 3.14159265358979323846f;
            float e  = __sinf(xr) * W_cog[d] + __cosf(xr) * W_cog[D + d] + b_cog[d];
            out[obase + 4 * D + d] = (cog == -1.0f) ? 0.0f : e;
        }
        // head: cyclical
        {
            float xr = head * deg2rad - 3.14159265358979323846f;
            float e  = __sinf(xr) * W_head[d] + __cosf(xr) * W_head[D + d] + b_head[d];
            out[obase + 5 * D + d] = (head == -1.0f) ? 0.0f : e;
        }
        // vtype: embedding gather
        {
            int vi = (int)vt;
            out[obase + 6 * D + d] = emb[vi * D + d];
        }
        // dr: continuous in [0, 30]
        {
            float xn = 2.0f * dr * (1.0f / 30.0f) - 1.0f;
            float e  = tanhf(xn * W_dr[d] + b_dr[d]);
            out[obase + 7 * D + d] = (dr == -1.0f) ? 0.0f : e;
        }
    }
}

extern "C" void kernel_launch(void* const* d_in, const int* in_sizes, int n_in,
                              void* d_out, int out_size, void* d_ws, size_t ws_size,
                              hipStream_t stream) {
    (void)in_sizes; (void)n_in; (void)out_size; (void)d_ws; (void)ws_size;
    const float* ais    = (const float*)d_in[0];
    const float* W_lon1 = (const float*)d_in[1];
    const float* b_lon1 = (const float*)d_in[2];
    const float* W_lon2 = (const float*)d_in[3];
    const float* b_lon2 = (const float*)d_in[4];
    const float* W_lat1 = (const float*)d_in[5];
    const float* b_lat1 = (const float*)d_in[6];
    const float* W_lat2 = (const float*)d_in[7];
    const float* b_lat2 = (const float*)d_in[8];
    const float* W_cog  = (const float*)d_in[9];
    const float* b_cog  = (const float*)d_in[10];
    const float* W_head = (const float*)d_in[11];
    const float* b_head = (const float*)d_in[12];
    const float* W_dr   = (const float*)d_in[13];
    const float* b_dr   = (const float*)d_in[14];
    const float* W_sog  = (const float*)d_in[15];
    const float* b_sog  = (const float*)d_in[16];
    const float* W_rot  = (const float*)d_in[17];
    const float* b_rot  = (const float*)d_in[18];
    const float* emb    = (const float*)d_in[19];
    float* out = (float*)d_out;

    dim3 grid(NTOK / TOKENS_PER_BLOCK);   // 256 blocks
    dim3 block(256);                      // 8 wave32
    hae_encoder_kernel<<<grid, block, 0, stream>>>(
        ais,
        W_lon1, b_lon1, W_lon2, b_lon2,
        W_lat1, b_lat1, W_lat2, b_lat2,
        W_cog, b_cog, W_head, b_head,
        W_dr, b_dr, W_sog, b_sog, W_rot, b_rot,
        emb, out);
}
